// MoDGATv2Layer_1116691497067
// MI455X (gfx1250) — matrix operand order repaired and checked by reading, not compiled, not run on hardware
//
#include <hip/hip_runtime.h>
#include <stdint.h>

#define NNODE 50000
#define NEDGE 800000
#define UNITS 128
#define DEPTHL 4
#define HEADS 8
#define HDIMH 16
#define EDIMF 64
#define EPSR 1e-6f

typedef unsigned short u16;
typedef __attribute__((ext_vector_type(8)))  __bf16 v8bf;
typedef __attribute__((ext_vector_type(16))) __bf16 v16bf;
typedef __attribute__((ext_vector_type(8)))  float  v8f;

// ---------- scalar helpers ----------
__device__ __forceinline__ float bf2f(u16 u) {
  union { float f; unsigned i; } v; v.i = ((unsigned)u) << 16; return v.f;
}
__device__ __forceinline__ u16 f2bf(float f) {
  union { float f; unsigned i; } v; v.f = f;
  unsigned r = v.i + 0x7fffu + ((v.i >> 16) & 1u);   // round-to-nearest-even
  return (u16)(r >> 16);
}
// order-preserving float<->int encoding for atomicMax-based segment max
__device__ __forceinline__ int encf(float f) {
  int i = __float_as_int(f);
  return (i < 0) ? (i ^ 0x7fffffff) : i;
}
__device__ __forceinline__ float decf(int e) {
  int i = (e < 0) ? (e ^ 0x7fffffff) : e;
  return __int_as_float(i);
}
__device__ __forceinline__ float wave_sum(float v) {
  #pragma unroll
  for (int m = 16; m > 0; m >>= 1) v += __shfl_xor(v, m, 32);
  return v;
}

// ---------- tiny utility kernels ----------
__global__ void fill_f32_kernel(float* p, float v, int n) {
  int t = blockIdx.x * blockDim.x + threadIdx.x;
  if (t < n) p[t] = v;
}
__global__ void fill_i32_kernel(int* p, int v, int n) {
  int t = blockIdx.x * blockDim.x + threadIdx.x;
  if (t < n) p[t] = v;
}
__global__ void cvt_f32_bf16_kernel(const float* __restrict__ s, u16* __restrict__ d, int n) {
  int t = blockIdx.x * blockDim.x + threadIdx.x;
  if (t < n) d[t] = f2bf(s[t]);
}
// W[K][Ncols] (f32, row-major) -> Wt[Ncols][K] (bf16)  : B operand wants contiguous K per column
__global__ void wtrans_kernel(const float* __restrict__ W, u16* __restrict__ Wt, int K, int Ncols) {
  int t = blockIdx.x * blockDim.x + threadIdx.x;
  if (t >= K * Ncols) return;
  int k = t / Ncols, n = t - k * Ncols;
  Wt[(size_t)n * K + k] = f2bf(W[t]);
}

// ---------- WMMA GEMM: C[M x 128] = A[M x K](bf16) * Wt[128 x K]^T(bf16) + bias ----------
// one wave = one 16x16 tile; 8 waves/block = full 128-wide strip; M % 16 == 0, K % 32 == 0
__global__ __launch_bounds__(256)
void gemm_bf16_kernel(const u16* __restrict__ A, const u16* __restrict__ Wt,
                      const float* __restrict__ bias, float* __restrict__ Cf,
                      u16* __restrict__ Cb, int M, int K) {
  const int lane  = threadIdx.x & 31;
  const int wave  = threadIdx.x >> 5;      // n-tile 0..7
  const int mtile = blockIdx.x;
  const int r     = lane & 15;
  const int half  = lane >> 4;

  const u16* arow = A  + (size_t)(mtile * 16 + r) * K;   // A: lane holds row r
  const u16* brow = Wt + (size_t)(wave * 16 + r) * K;    // B: lane holds column r

  v8f c = {0.f, 0.f, 0.f, 0.f, 0.f, 0.f, 0.f, 0.f};
  for (int kb = 0; kb < K; kb += 32) {
    // A 16x32 frag (ISA 7.12.2): lanes 0-15 K={0..7,16..23}, lanes 16-31 +8
    const int oa = kb + half * 8;
    v8bf alo = *(const v8bf*)(arow + oa);
    v8bf ahi = *(const v8bf*)(arow + oa + 16);
    // B 32x16 frag: same K order as A: lanes 0-15 K={0..7,16..23}, lanes 16-31 +8
    const int ob = kb + half * 8;
    v8bf blo = *(const v8bf*)(brow + ob);
    v8bf bhi = *(const v8bf*)(brow + ob + 16);
    v16bf af = __builtin_shufflevector(alo, ahi, 0,1,2,3,4,5,6,7,8,9,10,11,12,13,14,15);
    v16bf bf = __builtin_shufflevector(blo, bhi, 0,1,2,3,4,5,6,7,8,9,10,11,12,13,14,15);
    c = __builtin_amdgcn_wmma_f32_16x16x32_bf16(false, af, false, bf, (short)0, c, false, false);
  }

  const int col = wave * 16 + r;           // C/D: lane -> column r, VGPR j -> row j+8*half
  const float bv = bias ? bias[col] : 0.f;
  #pragma unroll
  for (int j = 0; j < 8; ++j) {
    const size_t m = (size_t)(mtile * 16 + j + 8 * half);
    const float v = c[j] + bv;
    if (Cf) Cf[m * UNITS + col] = v;
    if (Cb) Cb[m * UNITS + col] = f2bf(v);
  }
}

// ---------- edge attention: score + segment max ----------
__global__ void score_kernel(const int* __restrict__ idx, const u16* __restrict__ ee,
                             const float* __restrict__ hn, const float* __restrict__ att,
                             float* __restrict__ score, int* __restrict__ menc, int E) {
  int t = blockIdx.x * blockDim.x + threadIdx.x;
  if (t >= E * HEADS) return;
  const int e = t >> 3, hh = t & 7;
  const int tgt = idx[2 * e], src = idx[2 * e + 1];
  const float* hs = hn + (size_t)src * UNITS + hh * HDIMH;
  const float* ht = hn + (size_t)tgt * UNITS + hh * HDIMH;
  const u16*   ep = ee + (size_t)e   * UNITS + hh * HDIMH;
  const float* av = att + hh * HDIMH;
  float s = 0.f;
  #pragma unroll
  for (int k = 0; k < HDIMH; ++k) {
    float v = hs[k] + ht[k] + bf2f(ep[k]);
    v = (v > 0.f) ? v : v * 0.2f;            // leaky relu
    s += v * av[k];
  }
  score[t] = s;
  atomicMax(&menc[tgt * HEADS + hh], encf(s));
}
__global__ void maxfix_kernel(int* m, int n) {   // int-encoded -> float; -inf -> 0
  int t = blockIdx.x * blockDim.x + threadIdx.x;
  if (t >= n) return;
  int e = m[t];
  float v = (e == INT_MIN) ? 0.f : decf(e);
  ((float*)m)[t] = v;
}
__global__ void denom_kernel(const int* __restrict__ idx, const float* __restrict__ score,
                             const float* __restrict__ m, float* __restrict__ den, int E) {
  int t = blockIdx.x * blockDim.x + threadIdx.x;
  if (t >= E * HEADS) return;
  const int e = t >> 3, hh = t & 7;
  const int tgt = idx[2 * e];
  atomicAdd(&den[tgt * HEADS + hh], __expf(score[t] - m[tgt * HEADS + hh]));
}
__global__ void aggregate_kernel(const int* __restrict__ idx, const float* __restrict__ score,
                                 const float* __restrict__ m, const float* __restrict__ den,
                                 const float* __restrict__ hn, float* __restrict__ acc, int E) {
  int t = blockIdx.x * blockDim.x + threadIdx.x;
  if (t >= E * HEADS) return;
  const int e = t >> 3, hh = t & 7;
  const int tgt = idx[2 * e], src = idx[2 * e + 1];
  const float alpha = __expf(score[t] - m[tgt * HEADS + hh]) / (den[tgt * HEADS + hh] + 1e-9f);
  const float* hs = hn + (size_t)src * UNITS + hh * HDIMH;
  float* ap = acc + (size_t)tgt * UNITS + hh * HDIMH;
  #pragma unroll
  for (int k = 0; k < HDIMH; ++k) atomicAdd(&ap[k], alpha * hs[k]);
}

// ---------- residual + rms(gamma) -> reprs ; rms(final_gamma) -> next h ----------
__global__ __launch_bounds__(128)
void residual_rms_kernel(const float* __restrict__ tmp, const float* __restrict__ h,
                         const float* __restrict__ gamma, const float* __restrict__ fgamma,
                         float* __restrict__ reprF, u16* __restrict__ reprB,
                         float* __restrict__ hnext, int layer, int N) {
  const int wave = threadIdx.x >> 5, lane = threadIdx.x & 31;
  const int n = blockIdx.x * 4 + wave;
  if (n >= N) return;
  float x[4]; float ss = 0.f;
  #pragma unroll
  for (int i = 0; i < 4; ++i) {
    const int u = lane * 4 + i;
    x[i] = tmp[(size_t)n * UNITS + u] + h[(size_t)n * UNITS + u];
    ss += x[i] * x[i];
  }
  ss = wave_sum(ss);
  const float rs = rsqrtf(ss * (1.f / UNITS) + EPSR);
  float r1[4]; float s2 = 0.f;
  #pragma unroll
  for (int i = 0; i < 4; ++i) {
    const int u = lane * 4 + i;
    r1[i] = x[i] * rs * gamma[u];
    s2 += r1[i] * r1[i];
  }
  s2 = wave_sum(s2);
  const float rs2 = rsqrtf(s2 * (1.f / UNITS) + EPSR);
  #pragma unroll
  for (int i = 0; i < 4; ++i) {
    const int u = lane * 4 + i;
    const size_t o = ((size_t)n * DEPTHL + layer) * UNITS + u;
    reprF[o] = r1[i];
    reprB[o] = f2bf(r1[i]);
    if (hnext) hnext[(size_t)n * UNITS + u] = r1[i] * rs2 * fgamma[u];
  }
}

// ---------- depth-attention fusion + final rms ----------
__global__ __launch_bounds__(128)
void fuse_kernel(const float* __restrict__ attin, const float* __restrict__ reprF,
                 const float* __restrict__ mowW, const float* __restrict__ mowb,
                 const float* __restrict__ fgamma, float* __restrict__ out, int N) {
  const int wave = threadIdx.x >> 5, lane = threadIdx.x & 31;
  const int n = blockIdx.x * 4 + wave;
  if (n >= N) return;
  float logit[DEPTHL];
  #pragma unroll
  for (int d = 0; d < DEPTHL; ++d) {
    float s = 0.f;
    #pragma unroll
    for (int i = 0; i < 4; ++i) {
      const int u = lane * 4 + i;
      s += tanhf(attin[((size_t)n * DEPTHL + d) * UNITS + u]) * mowW[u];
    }
    logit[d] = wave_sum(s) + mowb[0];
  }
  float mx = logit[0];
  #pragma unroll
  for (int d = 1; d < DEPTHL; ++d) mx = fmaxf(mx, logit[d]);
  float w[DEPTHL], den = 0.f;
  #pragma unroll
  for (int d = 0; d < DEPTHL; ++d) { w[d] = __expf(logit[d] - mx); den += w[d]; }
  const float inv = 1.f / den;
  float fused[4]; float ss = 0.f;
  #pragma unroll
  for (int i = 0; i < 4; ++i) {
    const int u = lane * 4 + i;
    float f = 0.f;
    #pragma unroll
    for (int d = 0; d < DEPTHL; ++d)
      f += w[d] * inv * reprF[((size_t)n * DEPTHL + d) * UNITS + u];
    fused[i] = f; ss += f * f;
  }
  ss = wave_sum(ss);
  const float rs = rsqrtf(ss * (1.f / UNITS) + EPSR);
  #pragma unroll
  for (int i = 0; i < 4; ++i) {
    const int u = lane * 4 + i;
    out[(size_t)n * UNITS + u] = fused[i] * rs * fgamma[u];
  }
}

// ---------------- host orchestration ----------------
static inline int cdiv(long long a, long long b) { return (int)((a + b - 1) / b); }

extern "C" void kernel_launch(void* const* d_in, const int* in_sizes, int n_in,
                              void* d_out, int out_size, void* d_ws, size_t ws_size,
                              hipStream_t stream) {
  const float* node  = (const float*)d_in[0];
  const float* edgef = (const float*)d_in[1];
  const int*   idxF  = (const int*)d_in[2];
  const int*   idxB  = (const int*)d_in[3];
  const float* WnF = (const float*)d_in[4],  *WeF = (const float*)d_in[5];
  const float* bF  = (const float*)d_in[6],  *attF = (const float*)d_in[7];
  const float* WnB = (const float*)d_in[8],  *WeB = (const float*)d_in[9];
  const float* bB  = (const float*)d_in[10], *attB = (const float*)d_in[11];
  const float* Wo  = (const float*)d_in[12], *bo = (const float*)d_in[13];
  const float* gam = (const float*)d_in[14];
  const float* moW = (const float*)d_in[15], *mob = (const float*)d_in[16];
  const float* mowW = (const float*)d_in[17], *mowb = (const float*)d_in[18];
  const float* fgam = (const float*)d_in[19];

  const size_t N = NNODE, E = NEDGE, U = UNITS;
  char* ws = (char*)d_ws;
  size_t off = 0;
  auto take = [&](size_t bytes) { size_t o = off; off = (off + bytes + 255) & ~(size_t)255; return o; };
  float* h       = (float*)(ws + take(N * U * 4));
  u16*   hbf     = (u16*)  (ws + take(N * U * 2));      // also reused as acc_bf16
  float* hn      = (float*)(ws + take(N * U * 4));
  float* acc     = (float*)(ws + take(N * U * 4));
  float* tmp     = (float*)(ws + take(N * U * 4));
  float* reprF   = (float*)(ws + take(N * DEPTHL * U * 4));
  u16*   reprB   = (u16*)  (ws + take(N * DEPTHL * U * 2));
  int*   mbuf    = (int*)  (ws + take(N * HEADS * 4));
  float* den     = (float*)(ws + take(N * HEADS * 4));
  float* score   = (float*)(ws + take(E * HEADS * 4));
  u16*   wt      = (u16*)  (ws + take(128 * 128 * 2)); // transposed-weight scratch
  u16*   edgebf  = (u16*)  (ws + take(E * EDIMF * 2));
  // big region: ee (bf16, E*128) during layers, att_in (f32, N*4*128) at the end
  char*  big     = ws + take(E * U * 2 > N * DEPTHL * U * 4 ? E * U * 2 : N * DEPTHL * U * 4);
  u16*   ee      = (u16*)big;
  float* attin   = (float*)big;

  const int B256 = 256;
  hipMemcpyAsync(h, node, N * U * 4, hipMemcpyDeviceToDevice, stream);
  cvt_f32_bf16_kernel<<<cdiv(E * EDIMF, B256), B256, 0, stream>>>(edgef, edgebf, (int)(E * EDIMF));

  auto run_dir = [&](const int* idx, const float* Wn, const float* We,
                     const float* bb, const float* att) {
    // hn = h @ Wn
    wtrans_kernel<<<cdiv(128 * 128, B256), B256, 0, stream>>>(Wn, wt, 128, 128);
    gemm_bf16_kernel<<<(int)(N / 16), B256, 0, stream>>>(hbf, wt, nullptr, hn, nullptr, (int)N, 128);
    // ee = e @ We + b   (stored bf16: halves the only HBM-sized stream)
    wtrans_kernel<<<cdiv(64 * 128, B256), B256, 0, stream>>>(We, wt, 64, 128);
    gemm_bf16_kernel<<<(int)(E / 16), B256, 0, stream>>>(edgebf, wt, bb, nullptr, ee, (int)E, 64);
    // segment softmax + aggregation
    fill_i32_kernel<<<cdiv(N * HEADS, B256), B256, 0, stream>>>(mbuf, INT_MIN, (int)(N * HEADS));
    fill_f32_kernel<<<cdiv(N * HEADS, B256), B256, 0, stream>>>(den, 0.f, (int)(N * HEADS));
    score_kernel<<<cdiv(E * HEADS, B256), B256, 0, stream>>>(idx, ee, hn, att, score, mbuf, (int)E);
    maxfix_kernel<<<cdiv(N * HEADS, B256), B256, 0, stream>>>(mbuf, (int)(N * HEADS));
    denom_kernel<<<cdiv(E * HEADS, B256), B256, 0, stream>>>(idx, score, (const float*)mbuf, den, (int)E);
    aggregate_kernel<<<cdiv(E * HEADS, B256), B256, 0, stream>>>(idx, score, (const float*)mbuf, den, hn, acc, (int)E);
  };

  for (int i = 0; i < DEPTHL; ++i) {
    cvt_f32_bf16_kernel<<<cdiv(N * U, B256), B256, 0, stream>>>(h, hbf, (int)(N * U));
    fill_f32_kernel<<<cdiv(N * U, B256), B256, 0, stream>>>(acc, 0.f, (int)(N * U));
    run_dir(idxF, WnF + (size_t)i * 128 * 128, WeF + (size_t)i * 64 * 128,
            bF + (size_t)i * 128, attF + (size_t)i * 128);
    run_dir(idxB, WnB + (size_t)i * 128 * 128, WeB + (size_t)i * 64 * 128,
            bB + (size_t)i * 128, attB + (size_t)i * 128);
    // tmp = (f + bk) @ Wo + bo
    wtrans_kernel<<<cdiv(128 * 128, B256), B256, 0, stream>>>(Wo + (size_t)i * 128 * 128, wt, 128, 128);
    cvt_f32_bf16_kernel<<<cdiv(N * U, B256), B256, 0, stream>>>(acc, hbf, (int)(N * U));
    gemm_bf16_kernel<<<(int)(N / 16), B256, 0, stream>>>(hbf, wt, bo + (size_t)i * 128, tmp, nullptr, (int)N, 128);
    residual_rms_kernel<<<cdiv(N, 4), 128, 0, stream>>>(
        tmp, h, gam + (size_t)i * 128, fgam, reprF, reprB,
        (i < DEPTHL - 1) ? h : nullptr, i, (int)N);
  }

  // att_in = stacked @ mo_W + mo_b  (M = N*DEPTH rows of 128)
  wtrans_kernel<<<cdiv(128 * 128, B256), B256, 0, stream>>>(moW, wt, 128, 128);
  gemm_bf16_kernel<<<(int)(N * DEPTHL / 16), B256, 0, stream>>>(reprB, wt, mob, attin, nullptr,
                                                               (int)(N * DEPTHL), 128);
  fuse_kernel<<<cdiv(N, 4), 128, 0, stream>>>(attin, reprF, mowW, mowb, fgam, (float*)d_out, (int)N);
}
